// WindowAttention_56384330662296
// MI455X (gfx1250) — compile-verified, matrix operand order checked by reading, not run on hardware
//
#include <hip/hip_runtime.h>
#include <hip/hip_bf16.h>

typedef _Float16 half_t;
typedef __attribute__((ext_vector_type(16))) _Float16 v16h;
typedef __attribute__((ext_vector_type(8)))  _Float16 v8h;
typedef __attribute__((ext_vector_type(8)))  float    v8f;

// ---------------------------------------------------------------------------
// helpers
// ---------------------------------------------------------------------------
__device__ __forceinline__ v8f zero_v8f() {
    v8f z;
#pragma unroll
    for (int i = 0; i < 8; ++i) z[i] = 0.0f;
    return z;
}

// WMMA D = A(16x32 f16) * B(32x16 f16) + C(16x16 f32)
__device__ __forceinline__ v8f wmma16(v16h a, v16h b, v8f c) {
    return __builtin_amdgcn_wmma_f32_16x16x32_f16(false, a, false, b,
                                                  (short)0, c, false, false);
}

// Load an A- or B-fragment of a 16x32 f16 tile from row-major storage.
// Lane mapping (identical for A and B frags of v_wmma_f32_16x16x32_f16):
//   lanes 0-15 : row = li, k in {k0..k0+7, k0+16..k0+23}
//   lanes 16-31: row = li, k in {k0+8..k0+15, k0+24..k0+31}
// -> two contiguous 16-byte chunks at (k0 + 8*lg) and (+16).
__device__ __forceinline__ v16h load_frag_h(const half_t* __restrict__ base,
                                            int rowStride, int row, int k0, int lg) {
    const half_t* p = base + row * rowStride + k0 + 8 * lg;
    v8h lo = *(const v8h*)(p);
    v8h hi = *(const v8h*)(p + 16);
    v16h a;
#pragma unroll
    for (int i = 0; i < 8; ++i) { a[i] = lo[i]; a[8 + i] = hi[i]; }
    return a;
}

// Same fragment but converting from f32 source on the fly.
__device__ __forceinline__ v16h load_frag_f32(const float* __restrict__ base,
                                              int rowStride, int row, int k0, int lg) {
    const float* p = base + row * rowStride + k0 + 8 * lg;
    v16h a;
#pragma unroll
    for (int i = 0; i < 8; ++i) {
        a[i]     = (half_t)p[i];
        a[8 + i] = (half_t)p[16 + i];
    }
    return a;
}

// ---------------------------------------------------------------------------
// Kernel 1: DynamicPosBias MLP  -> pos[961][6]
// ---------------------------------------------------------------------------
__device__ __forceinline__ void ln_relu12(float* x, const float* g, const float* b) {
    float m = 0.f;
#pragma unroll
    for (int j = 0; j < 12; ++j) m += x[j];
    m *= (1.0f / 12.0f);
    float v = 0.f;
#pragma unroll
    for (int j = 0; j < 12; ++j) { float d = x[j] - m; v += d * d; }
    v *= (1.0f / 12.0f);
    float inv = rsqrtf(v + 1e-5f);
#pragma unroll
    for (int j = 0; j < 12; ++j) {
        float t = (x[j] - m) * inv * g[j] + b[j];
        x[j] = fmaxf(t, 0.0f);
    }
}

__global__ void dpb_mlp_kernel(const float* __restrict__ w_pp, const float* __restrict__ b_pp,
                               const float* __restrict__ g1, const float* __restrict__ bb1,
                               const float* __restrict__ w1, const float* __restrict__ b1,
                               const float* __restrict__ g2, const float* __restrict__ bb2,
                               const float* __restrict__ w2, const float* __restrict__ b2,
                               const float* __restrict__ g3, const float* __restrict__ bb3,
                               const float* __restrict__ w3, const float* __restrict__ b3,
                               float* __restrict__ pos) {
    int i = blockIdx.x * blockDim.x + threadIdx.x;
    if (i >= 961) return;  // (2*16-1)^2
    float dh = (float)(i / 31 - 15);
    float dw = (float)(i % 31 - 15);
    float x[12], y[12];
#pragma unroll
    for (int j = 0; j < 12; ++j) x[j] = dh * w_pp[j] + dw * w_pp[12 + j] + b_pp[j];
    ln_relu12(x, g1, bb1);
#pragma unroll
    for (int j = 0; j < 12; ++j) {
        float s = b1[j];
#pragma unroll
        for (int k = 0; k < 12; ++k) s += x[k] * w1[k * 12 + j];
        y[j] = s;
    }
    ln_relu12(y, g2, bb2);
#pragma unroll
    for (int j = 0; j < 12; ++j) {
        float s = b2[j];
#pragma unroll
        for (int k = 0; k < 12; ++k) s += y[k] * w2[k * 12 + j];
        x[j] = s;
    }
    ln_relu12(x, g3, bb3);
#pragma unroll
    for (int h = 0; h < 6; ++h) {
        float s = b3[h];
#pragma unroll
        for (int k = 0; k < 12; ++k) s += x[k] * w3[k * 6 + h];
        pos[i * 6 + h] = s;
    }
}

// ---------------------------------------------------------------------------
// Kernel 2: gather both bias tables into [h][n][m] layout (rpi == idx)
// ---------------------------------------------------------------------------
__global__ void make_bias_kernel(const float* __restrict__ table,
                                 const float* __restrict__ pos,
                                 const int* __restrict__ rpi,
                                 float* __restrict__ biasA,
                                 float* __restrict__ biasB) {
    int i = blockIdx.x * 256 + threadIdx.x;  // 0..65535  (n*256+m)
    int r = rpi[i];
#pragma unroll
    for (int h = 0; h < 6; ++h) {
        biasA[h * 65536 + i] = table[r * 6 + h];
        biasB[h * 65536 + i] = pos[r * 6 + h];
    }
}

// ---------------------------------------------------------------------------
// Kernel 3: QKV projection GEMM.
// X[65536,192] f32 @ W[192,384] f32 + bias -> s=0 half scaled to QK buffer
// (row-major [b][h][n][32] f16), s=1 half to transposed V buffer
// ([b][h][32][256] f16).  Block tile: 128 rows x 64 cols, 8 waves.
// ---------------------------------------------------------------------------
__global__ __launch_bounds__(256) void gemm_qkv_kernel(
    const float* __restrict__ X, const float* __restrict__ W,
    const float* __restrict__ bias, half_t* __restrict__ outQK,
    half_t* __restrict__ outVT, float scale_s0) {
    __shared__ half_t wl[64 * 192];  // transposed weight slab: wl[cl*192+k]
    int tid = threadIdx.x;
    int c0 = blockIdx.y * 64;
    for (int i = tid; i < 64 * 192; i += 256) {
        int cl = i / 192, k = i - cl * 192;
        wl[i] = (half_t)W[k * 384 + c0 + cl];
    }
    __syncthreads();

    int wave = tid >> 5, lane = tid & 31, lg = lane >> 4, li = lane & 15;
    int m0 = blockIdx.x * 128 + wave * 16;

    v8f acc[4];
#pragma unroll
    for (int t = 0; t < 4; ++t) acc[t] = zero_v8f();

    for (int k0 = 0; k0 < 192; k0 += 32) {
        v16h a = load_frag_f32(X, 192, m0 + li, k0, lg);
#pragma unroll
        for (int t = 0; t < 4; ++t) {
            v16h b = load_frag_h(wl, 192, t * 16 + li, k0, lg);
            acc[t] = wmma16(a, b, acc[t]);
        }
    }

#pragma unroll
    for (int t = 0; t < 4; ++t) {
        int cg = c0 + t * 16 + li;
        float bb = bias[cg];
        int s = cg / 192;
        int rem = cg - s * 192;
        int h = rem >> 5, dd = rem & 31;
#pragma unroll
        for (int r = 0; r < 8; ++r) {
            int rowg = m0 + r + 8 * lg;
            int b_ = rowg >> 8, n = rowg & 255;
            float v = acc[t][r] + bb;
            if (s == 0) {
                outQK[(((b_ * 6 + h) * 256 + n) * 32) + dd] = (half_t)(v * scale_s0);
            } else {
                outVT[(((b_ * 6 + h) * 32 + dd) * 256) + n] = (half_t)v;
            }
        }
    }
}

// ---------------------------------------------------------------------------
// Kernel 4: fused dual-bias attention.  One block (8 waves) per (b,h).
// Each wave: 2 row tiles of 16; per row tile: 16 QK WMMAs (d=32 = one K step),
// exact softmax over 256 cols in registers (shfl reductions), exp -> per-wave
// LDS P slab, then 16 PV WMMAs per path reading V^T fragments from global.
// Bias values are re-read in the exp pass (they live in L2: 3MB total vs
// 192MB L2) instead of being cached in 128 extra VGPRs -> ~2x occupancy.
// ---------------------------------------------------------------------------
__global__ __launch_bounds__(256) void attn_kernel(
    const half_t* __restrict__ Q, const half_t* __restrict__ K,
    const half_t* __restrict__ VT, const half_t* __restrict__ VHT,
    const float* __restrict__ biasA, const float* __restrict__ biasB,
    half_t* __restrict__ O1, half_t* __restrict__ O2) {
    __shared__ half_t P[8][16 * 256];  // 8 waves x 8KB = 64KB
    int bh = blockIdx.x;               // 0..1535
    int b = bh / 6, h = bh - b * 6;
    int tid = threadIdx.x, wave = tid >> 5, lane = tid & 31;
    int lg = lane >> 4, li = lane & 15;

    const half_t* Qb = Q + (size_t)bh * 256 * 32;
    const half_t* Kb = K + (size_t)bh * 256 * 32;
    const half_t* V0 = VT + (size_t)bh * 32 * 256;
    const half_t* V1 = VHT + (size_t)bh * 32 * 256;
    const float* bA = biasA + (size_t)h * 65536;
    const float* bB = biasB + (size_t)h * 65536;
    half_t* myP = &P[wave][0];

    for (int rt = 0; rt < 2; ++rt) {
        int n0 = wave * 32 + rt * 16;
        v16h qf = load_frag_h(Qb, 32, n0 + li, 0, lg);

        v8f acc[16];  // full 16x256 score block for these 16 rows
#pragma unroll
        for (int t = 0; t < 16; ++t) {
            v16h kf = load_frag_h(Kb, 32, t * 16 + li, 0, lg);
            acc[t] = wmma16(qf, kf, zero_v8f());
        }

#pragma unroll 1
        for (int path = 0; path < 2; ++path) {
            const float* bias = path ? bB : bA;
            const half_t* Vt = path ? V1 : V0;
            half_t* Ob = path ? O2 : O1;

            // pass 1: row max of (acc + bias)
            float m[8];
#pragma unroll
            for (int r = 0; r < 8; ++r) m[r] = -3.0e30f;
#pragma unroll
            for (int t = 0; t < 16; ++t) {
#pragma unroll
                for (int r = 0; r < 8; ++r) {
                    float bv = bias[(n0 + r + 8 * lg) * 256 + t * 16 + li];
                    m[r] = fmaxf(m[r], acc[t][r] + bv);
                }
            }
#pragma unroll
            for (int mask = 1; mask < 16; mask <<= 1)
#pragma unroll
                for (int r = 0; r < 8; ++r)
                    m[r] = fmaxf(m[r], __shfl_xor(m[r], mask, 32));

            // pass 2: exp -> LDS (C layout -> row-major), row sums
            float l[8];
#pragma unroll
            for (int r = 0; r < 8; ++r) l[r] = 0.f;
#pragma unroll
            for (int t = 0; t < 16; ++t) {
#pragma unroll
                for (int r = 0; r < 8; ++r) {
                    float bv = bias[(n0 + r + 8 * lg) * 256 + t * 16 + li];
                    float e = __expf(acc[t][r] + bv - m[r]);
                    l[r] += e;
                    myP[(r + 8 * lg) * 256 + t * 16 + li] = (half_t)e;
                }
            }
#pragma unroll
            for (int mask = 1; mask < 16; mask <<= 1)
#pragma unroll
                for (int r = 0; r < 8; ++r) l[r] += __shfl_xor(l[r], mask, 32);

            // PV: o[16x32] = P[16x256] @ V[256x32]; V stored transposed so
            // B-frag reads are contiguous.
            v8f o0 = zero_v8f(), o1 = zero_v8f();
#pragma unroll
            for (int kt = 0; kt < 8; ++kt) {
                v16h pf = load_frag_h(myP, 256, li, kt * 32, lg);
                v16h bf0 = load_frag_h(Vt, 256, li, kt * 32, lg);
                v16h bf1 = load_frag_h(Vt + 16 * 256, 256, li, kt * 32, lg);
                o0 = wmma16(pf, bf0, o0);
                o1 = wmma16(pf, bf1, o1);
            }

            // normalize and write O in [b*N, C] f16 layout for the proj GEMM
#pragma unroll
            for (int r = 0; r < 8; ++r) {
                float inv = 1.0f / l[r];
                int ng = b * 256 + n0 + r + 8 * lg;
                Ob[(size_t)ng * 192 + h * 32 + li] = (half_t)(o0[r] * inv);
                Ob[(size_t)ng * 192 + h * 32 + 16 + li] = (half_t)(o1[r] * inv);
            }
        }
    }
}

// ---------------------------------------------------------------------------
// Kernel 5: output projection GEMM.  O[65536,192] f16 @ W[192,192] + b -> f32
// ---------------------------------------------------------------------------
__global__ __launch_bounds__(256) void gemm_proj_kernel(
    const half_t* __restrict__ O, const float* __restrict__ W,
    const float* __restrict__ bias, float* __restrict__ Y) {
    __shared__ half_t wl[64 * 192];
    int tid = threadIdx.x;
    int c0 = blockIdx.y * 64;
    for (int i = tid; i < 64 * 192; i += 256) {
        int cl = i / 192, k = i - cl * 192;
        wl[i] = (half_t)W[k * 192 + c0 + cl];
    }
    __syncthreads();

    int wave = tid >> 5, lane = tid & 31, lg = lane >> 4, li = lane & 15;
    int m0 = blockIdx.x * 128 + wave * 16;

    v8f acc[4];
#pragma unroll
    for (int t = 0; t < 4; ++t) acc[t] = zero_v8f();

    for (int k0 = 0; k0 < 192; k0 += 32) {
        v16h a = load_frag_h(O, 192, m0 + li, k0, lg);
#pragma unroll
        for (int t = 0; t < 4; ++t) {
            v16h b = load_frag_h(wl, 192, t * 16 + li, k0, lg);
            acc[t] = wmma16(a, b, acc[t]);
        }
    }

#pragma unroll
    for (int t = 0; t < 4; ++t) {
        int cg = c0 + t * 16 + li;
        float bb = bias[cg];
#pragma unroll
        for (int r = 0; r < 8; ++r) {
            int rowg = m0 + r + 8 * lg;
            Y[(size_t)rowg * 192 + cg] = acc[t][r] + bb;
        }
    }
}

// ---------------------------------------------------------------------------
// launch
// ---------------------------------------------------------------------------
extern "C" void kernel_launch(void* const* d_in, const int* in_sizes, int n_in,
                              void* d_out, int out_size, void* d_ws, size_t ws_size,
                              hipStream_t stream) {
    const float* x1 = (const float*)d_in[0];
    const float* x2 = (const float*)d_in[1];
    const float* table = (const float*)d_in[2];
    const float* w_qv = (const float*)d_in[3];
    const float* b_qv = (const float*)d_in[4];
    const float* w_kv = (const float*)d_in[5];
    const float* b_kv = (const float*)d_in[6];
    const float* w_p1 = (const float*)d_in[7];
    const float* b_p1 = (const float*)d_in[8];
    const float* w_p2 = (const float*)d_in[9];
    const float* b_p2 = (const float*)d_in[10];
    const float* w_pp = (const float*)d_in[11];
    const float* b_pp = (const float*)d_in[12];
    const float* ln1g = (const float*)d_in[13];
    const float* ln1b = (const float*)d_in[14];
    const float* w1 = (const float*)d_in[15];
    const float* b1 = (const float*)d_in[16];
    const float* ln2g = (const float*)d_in[17];
    const float* ln2b = (const float*)d_in[18];
    const float* w2 = (const float*)d_in[19];
    const float* b2 = (const float*)d_in[20];
    const float* ln3g = (const float*)d_in[21];
    const float* ln3b = (const float*)d_in[22];
    const float* w3 = (const float*)d_in[23];
    const float* b3 = (const float*)d_in[24];
    const int* rpi = (const int*)d_in[25];
    (void)in_sizes; (void)n_in; (void)out_size; (void)ws_size;

    // workspace carve-up
    const size_t SZ_QKV = (size_t)256 * 6 * 256 * 32 * sizeof(half_t);  // 25,165,824
    char* ws = (char*)d_ws;
    half_t* Qh = (half_t*)(ws);                 ws += SZ_QKV;
    half_t* Kh = (half_t*)(ws);                 ws += SZ_QKV;
    half_t* VTh = (half_t*)(ws);                ws += SZ_QKV;
    half_t* VHTh = (half_t*)(ws);               ws += SZ_QKV;
    half_t* O1h = (half_t*)(ws);                ws += SZ_QKV;
    half_t* O2h = (half_t*)(ws);                ws += SZ_QKV;
    float* biasA = (float*)(ws);                ws += (size_t)6 * 65536 * sizeof(float);
    float* biasB = (float*)(ws);                ws += (size_t)6 * 65536 * sizeof(float);
    float* pos = (float*)(ws);                  ws += 961 * 6 * sizeof(float);

    float* y1 = (float*)d_out;
    float* y2 = y1 + (size_t)65536 * 192;

    const float scale = 0.17677669529663689f;  // 32^-0.5

    dpb_mlp_kernel<<<4, 256, 0, stream>>>(w_pp, b_pp, ln1g, ln1b, w1, b1,
                                          ln2g, ln2b, w2, b2, ln3g, ln3b,
                                          w3, b3, pos);
    make_bias_kernel<<<256, 256, 0, stream>>>(table, pos, rpi, biasA, biasB);

    gemm_qkv_kernel<<<dim3(512, 6), 256, 0, stream>>>(x1, w_qv, b_qv, Qh, VTh, scale);
    gemm_qkv_kernel<<<dim3(512, 6), 256, 0, stream>>>(x2, w_kv, b_kv, Kh, VHTh, 1.0f);

    attn_kernel<<<1536, 256, 0, stream>>>(Qh, Kh, VTh, VHTh, biasA, biasB, O1h, O2h);

    gemm_proj_kernel<<<dim3(512, 3), 256, 0, stream>>>(O1h, w_p1, b_p1, y1);
    gemm_proj_kernel<<<dim3(512, 3), 256, 0, stream>>>(O2h, w_p2, b_p2, y2);
}
